// Smoother_41128606826509
// MI455X (gfx1250) — matrix operand order repaired and checked by reading, not compiled, not run on hardware
//
#include <hip/hip_runtime.h>

// CDNA5 / gfx1250 WMMA fragment types
typedef __attribute__((ext_vector_type(16))) _Float16 v16h;
typedef __attribute__((ext_vector_type(8)))  float    v8f;

#define T_DIM 2048
#define B_DIM 32
#define ROWS_PER_BLOCK 64   // 4 waves x 16 rows
#define BLOCK_THREADS 128
#define SQRT_2PI 2.5066282746310002f
#define LOG2E    1.4426950408889634f

__launch_bounds__(BLOCK_THREADS)
__global__ void smoother_wmma_kernel(const float* __restrict__ src,
                                     const float* __restrict__ timeA,
                                     const float* __restrict__ mask,
                                     const float* __restrict__ p_ptr,
                                     float* __restrict__ out) {
    // Per-batch staging in LDS (one batch per block)
    __shared__ __align__(16) float    gS[T_DIM];    // g[s] = exp(-time[s]/(2p))*mask[s]  (8 KB)
    __shared__ __align__(32) _Float16 srcH[T_DIM];  // src[s] as f16 for the B matrix     (4 KB)
    __shared__ float redMax[BLOCK_THREADS];
    __shared__ float redMin[BLOCK_THREADS];
    __shared__ float dOut[4][32][8];                // per-wave accumulator spill         (4 KB)

    const int b    = blockIdx.y;
    const int t0bk = blockIdx.x * ROWS_PER_BLOCK;
    const int tid  = threadIdx.x;

    const float p     = p_ptr[0];
    const float C     = 1.0f / (p * SQRT_2PI);
    const float negHp = -0.5f / p;
    const float posHp =  0.5f / p;

    // ---- Phase 1: build g[s], srcH[s]; reduce gmax/gmin over all sources ----
    float lmax = -__builtin_inff();
    float lmin =  __builtin_inff();
    for (int s = tid; s < T_DIM; s += BLOCK_THREADS) {
        float ts = timeA[s * B_DIM + b];
        float g  = __expf(negHp * ts) * mask[s * B_DIM + b];
        gS[s]    = g;
        srcH[s]  = (_Float16)src[s * B_DIM + b];
        lmax = fmaxf(lmax, g);
        lmin = fminf(lmin, g);
    }
    redMax[tid] = lmax;
    redMin[tid] = lmin;
    __syncthreads();
    for (int off = BLOCK_THREADS / 2; off > 0; off >>= 1) {
        if (tid < off) {
            redMax[tid] = fmaxf(redMax[tid], redMax[tid + off]);
            redMin[tid] = fminf(redMin[tid], redMin[tid + off]);
        }
        __syncthreads();
    }
    const float gmax = redMax[0];
    const float gmin = redMin[0];

    // ---- Phase 2: flash-style loop, one wave per 16-center tile ----
    const int wave = tid >> 5;
    const int lane = tid & 31;
    const int m    = lane & 15;                  // matrix row (center) this lane serves
    const int t    = t0bk + wave * 16 + m;

    // rank-1 logits: coef[t,s] = a_m * g[s]; exact row max from gmax/gmin.
    // Fold log2(e) into the constants so each element is one FMA + one v_exp_f32.
    const float a_m  = C * __expf(posHp * timeA[t * B_DIM + b]);
    const float mx_m = (a_m >= 0.0f) ? a_m * gmax : a_m * gmin;
    const float a2   = a_m  * LOG2E;
    const float mx2  = mx_m * LOG2E;

    // A-matrix (16x32) k order used here (contiguous, as B): lanes 0-15 hold K=0..15,
    // lanes 16-31 hold K=16..31 for row M = lane&15.
    const int kbaseA = (lane < 16) ? 0 : 16;
    // B-matrix (32x16) k order used here (contiguous, as A): lanes 0-15 hold K=0..15 (N=lane),
    // lanes 16-31 hold K=16..31 (N=lane-16).
    const int kbaseB = (lane < 16) ? 0 : 16;
    const int n      = m;                        // B-matrix column for this lane

    // Loop-invariant branchless selection masks: column 0 = src, column 1 = ones.
    const _Float16 selSrc = (n == 0) ? (_Float16)1.0f : (_Float16)0.0f;
    const _Float16 selOne = (n == 1) ? (_Float16)1.0f : (_Float16)0.0f;

    v8f acc = {};                                // col0 -> numerator, col1 -> denominator

#pragma unroll 2
    for (int s0 = 0; s0 < T_DIM; s0 += 32) {
        // ---- A fragment: P = exp2(a2*g - mx2), batched so the 16 v_exp_f32
        //      issue back-to-back (fills all TRANS32 hazard slots). ----
        float logit[16];
#pragma unroll
        for (int j = 0; j < 16; ++j) {
            const int k = kbaseA + j;
            logit[j] = a2 * gS[s0 + k] - mx2;    // LDS broadcast read + FMA
        }
        float pe[16];
#pragma unroll
        for (int j = 0; j < 16; ++j) {
            pe[j] = __builtin_amdgcn_exp2f(logit[j]);   // raw v_exp_f32
        }
        v16h afrag;
#pragma unroll
        for (int j = 0; j < 16; ++j) {
            afrag[j] = (_Float16)pe[j];          // pairs fuse into v_cvt_pk_f16_f32
        }

        // ---- B fragment, branch-free: one 32B vector LDS load (broadcast within
        //      each half-wave) masked by per-lane constants -> packed f16 FMAs. ----
        const v16h sv = *(const v16h*)(srcH + s0 + kbaseB);
        const v16h bfrag = sv * selSrc + selOne;

        // D = A x B + C  : v_wmma_f32_16x16x32_f16
        acc = __builtin_amdgcn_wmma_f32_16x16x32_f16(
                  /*neg_a=*/false, afrag, /*neg_b=*/false, bfrag,
                  /*c_mod=*/(short)0, acc, /*reuse_a=*/false, /*reuse_b=*/false);
    }

    // ---- Phase 3: extract num/den from the C/D layout and finish ----
    // C/D 16x16 f32: VGPR r, lanes 0-15 -> (M=r, N=lane); lanes 16-31 -> (M=r+8, N=lane-16)
#pragma unroll
    for (int r = 0; r < 8; ++r) dOut[wave][lane][r] = acc[r];
    __syncthreads();

    if (lane < 16) {
        const int srcLane = (m < 8) ? 0 : 16;    // N==0 column lives in lane 0 / lane 16
        const int r       = m & 7;
        const float num = dOut[wave][srcLane][r];
        const float den = dOut[wave][srcLane + 1][r];
        out[t * B_DIM + b] = num / den;
    }
}

extern "C" void kernel_launch(void* const* d_in, const int* in_sizes, int n_in,
                              void* d_out, int out_size, void* d_ws, size_t ws_size,
                              hipStream_t stream) {
    const float* src   = (const float*)d_in[0];  // (T,B,1)
    const float* timeA = (const float*)d_in[1];  // (T,B)
    const float* mask  = (const float*)d_in[2];  // (T,B,1)
    const float* p     = (const float*)d_in[3];  // (1,)
    float* out = (float*)d_out;                  // (T,B,1)

    dim3 grid(T_DIM / ROWS_PER_BLOCK, B_DIM);    // 32 x 32 blocks
    smoother_wmma_kernel<<<grid, BLOCK_THREADS, 0, stream>>>(src, timeA, mask, p, out);
}
